// AttnPairBiasFullWrapper_83708912599800
// MI455X (gfx1250) — compile-verified, matrix operand order checked by reading, not run on hardware
//
#include <hip/hip_runtime.h>
#include <hip/hip_bf16.h>
#include <math.h>

// ---------------------------------------------------------------------------
// Problem dims (fixed by reference): B=1, N=512, C_S=384, C_Z=128, H=16, D=24
// ---------------------------------------------------------------------------
#define N_SEQ 512
#define CS    384
#define CZ    128
#define NH    16
#define DH    24
#define DP    32          // D padded to 32 for K=32 WMMA
#define EPS   1e-5f

typedef __attribute__((ext_vector_type(16))) _Float16 v16h;
typedef __attribute__((ext_vector_type(8)))  float    v8f;
typedef __attribute__((ext_vector_type(4)))  unsigned u32x4;
typedef __attribute__((ext_vector_type(8)))  unsigned u32x8;

__device__ __forceinline__ v8f wmma32(v16h a, v16h b, v8f c) {
  // D = A(16x32 f16) * B(32x16 f16) + C(16x16 f32)
  return __builtin_amdgcn_wmma_f32_16x16x32_f16(false, a, false, b, (short)0, c,
                                                false, false);
}

// A-frag / Bt-frag loader: 16 contiguous halves per lane, 32B aligned.
__device__ __forceinline__ v16h load_frag(const _Float16* __restrict__ base,
                                          int stride, int row, int col) {
  int lane = threadIdx.x & 31;
  return *(const v16h*)(base + (size_t)(row + (lane & 15)) * stride + col +
                        ((lane >> 4) << 4));
}

// LDS byte offset of a __shared__ object (generic -> AS(3) -> int).
__device__ __forceinline__ unsigned lds_off(const void* p) {
  return (unsigned)(unsigned long long)(__attribute__((address_space(3))) const char*)p;
}

// Tensor Data Mover: DMA `ndw` contiguous DWORDs from global memory into LDS.
// 1-D D# descriptor: data_size=4B, tile_dim0 = tensor_dim0 = ndw, no padding,
// no multicast (workgroup_mask = 0), user-mode descriptor (count=1, type=2).
// Tracked with TENSORcnt; pair with __builtin_amdgcn_s_wait_tensorcnt.
__device__ __forceinline__ void tdm_load_1d(unsigned lds_byte_off,
                                            const void* gsrc, unsigned ndw) {
  unsigned long long ga = (unsigned long long)gsrc;
  u32x4 g0;
  g0[0] = 1u;                                                // count=1 (valid)
  g0[1] = lds_byte_off;                                      // lds_addr [63:32]
  g0[2] = (unsigned)ga;                                      // global_addr[31:0]
  g0[3] = (unsigned)((ga >> 32) & 0x01FFFFFFu) | (2u << 30); // addr[56:32], type=2
  u32x8 g1;
  g1[0] = 2u << 16;                       // workgroup_mask=0, data_size=2 (4B)
  g1[1] = (ndw & 0xFFFFu) << 16;          // tensor_dim0[15:0]  (bits 63:48)
  g1[2] = ((ndw >> 16) & 0xFFFFu) | (1u << 16); // tensor_dim0[31:16], tensor_dim1=1
  g1[3] = (ndw & 0xFFFFu) << 16;          // tile_dim0 = ndw    (bits 127:112)
  g1[4] = 0;                              // tile_dim1=0, tile_dim2=0 (1-D tile)
  g1[5] = ndw;                            // tensor_dim0_stride[31:0]
  g1[6] = 0;
  g1[7] = 0;
  asm volatile("tensor_load_to_lds %0, %1" :: "s"(g0), "s"(g1) : "memory");
}

// ---------------------------------------------------------------------------
// Kernel 1: LayerNorm(a) -> f16   (512 rows x 384)
// ---------------------------------------------------------------------------
__global__ void k_ln_a(const float* __restrict__ a, const float* __restrict__ g,
                       const float* __restrict__ b, _Float16* __restrict__ out) {
  int row = blockIdx.x;
  int t = threadIdx.x;              // 128 threads = 4 waves
  const float* x = a + (size_t)row * CS;
  float s = 0.f, ss = 0.f;
  for (int i = t; i < CS; i += 128) { float v = x[i]; s += v; ss += v * v; }
  #pragma unroll
  for (int m = 16; m >= 1; m >>= 1) {
    s += __shfl_xor(s, m, 32);
    ss += __shfl_xor(ss, m, 32);
  }
  __shared__ float rs[4], rss[4];
  if ((t & 31) == 0) { rs[t >> 5] = s; rss[t >> 5] = ss; }
  __syncthreads();
  s = rs[0] + rs[1] + rs[2] + rs[3];
  ss = rss[0] + rss[1] + rss[2] + rss[3];
  float mean = s * (1.f / CS);
  float var = ss * (1.f / CS) - mean * mean;
  float inv = rsqrtf(var + EPS);
  for (int i = t; i < CS; i += 128)
    out[(size_t)row * CS + i] = (_Float16)(((x[i] - mean) * inv) * g[i] + b[i]);
}

// ---------------------------------------------------------------------------
// Kernel 2: weight conversion f32 -> f16, transposed to [n][k] for Bt frags
// ---------------------------------------------------------------------------
__global__ void k_prep_w(const float* __restrict__ Wq, const float* __restrict__ Wk,
                         const float* __restrict__ Wv, const float* __restrict__ Wg,
                         const float* __restrict__ Wo, const float* __restrict__ Wz,
                         _Float16* __restrict__ WqT, _Float16* __restrict__ WkT,
                         _Float16* __restrict__ WvT, _Float16* __restrict__ WgT,
                         _Float16* __restrict__ WoT, _Float16* __restrict__ WzT) {
  int stride = gridDim.x * blockDim.x;
  int idx0 = blockIdx.x * blockDim.x + threadIdx.x;
  for (int i = idx0; i < CS * CS; i += stride) {
    int k = i / CS, n = i % CS;     // source W[k][n] -> WT[n][k]
    WqT[(size_t)n * CS + k] = (_Float16)Wq[i];
    WkT[(size_t)n * CS + k] = (_Float16)Wk[i];
    WvT[(size_t)n * CS + k] = (_Float16)Wv[i];
    WgT[(size_t)n * CS + k] = (_Float16)Wg[i];
    WoT[(size_t)n * CS + k] = (_Float16)Wo[i];
  }
  for (int i = idx0; i < CZ * NH; i += stride) {
    int k = i / NH, n = i % NH;
    WzT[n * CZ + k] = (_Float16)Wz[i];
  }
}

// ---------------------------------------------------------------------------
// Kernel 3: generic f16 WMMA GEMM: C[M][N] = A[M][K] * Bt[N][K]^T (+ bias)
// block = 128 threads (4 waves), wave -> one 16x16 tile; grid (M/16, N/64)
// ---------------------------------------------------------------------------
__global__ void k_gemm_f16(const _Float16* __restrict__ A,
                           const _Float16* __restrict__ Bt,
                           const float* __restrict__ bias,
                           float* __restrict__ C, int M, int N, int K) {
  int wave = threadIdx.x >> 5;
  int lane = threadIdx.x & 31;
  int m0 = blockIdx.x * 16;
  int n0 = blockIdx.y * 64 + wave * 16;
  if (m0 >= M || n0 >= N) return;
  v8f acc = {};
  for (int k0 = 0; k0 < K; k0 += 32) {
    v16h af = load_frag(A, K, m0, k0);
    v16h bf = load_frag(Bt, K, n0, k0);
    acc = wmma32(af, bf, acc);
  }
  int col = n0 + (lane & 15);
  float bv = bias ? bias[col] : 0.f;
  #pragma unroll
  for (int v = 0; v < 8; ++v) {
    int row = m0 + v + ((lane >> 4) << 3);   // D-layout: VGPR v -> row v (+8 hi half)
    C[(size_t)row * N + col] = acc[v] + bv;
  }
}

// ---------------------------------------------------------------------------
// Kernel 4: fused LN(z) + pair-bias projection, TDM double-buffered.
// One wave per block; block handles (i, 4 consecutive 16-j tiles).
//   bias[i][j][h] = LN(z[i,j,:]) @ Wz[:,h] + bz[h]
// Each 16x128 f32 z-tile (8 KB, fully contiguous) is DMA'd into LDS by the
// Tensor Data Mover while the previous tile is normalized + WMMA'd.
// ---------------------------------------------------------------------------
__global__ void k_pair_bias(const float* __restrict__ z, const float* __restrict__ gz,
                            const float* __restrict__ bz_ln,
                            const _Float16* __restrict__ WzT,
                            const float* __restrict__ bz,
                            float* __restrict__ bias) {
  int i = blockIdx.x;
  int jblk = blockIdx.y;            // 8 blocks x 4 tiles = 32 tiles of 16 j
  int lane = threadIdx.x;           // 32 threads (1 wave)

  __shared__ float    zstage[2][16 * CZ];   // 2 x 8KB TDM staging buffers
  __shared__ _Float16 zt[16][CZ];           // normalized f16 tile for WMMA

  // LN affine params, register-resident (lane l owns channels l,l+32,l+64,l+96)
  float gzc[4], bzc[4];
  #pragma unroll
  for (int c = 0; c < 4; ++c) { gzc[c] = gz[lane + 32 * c]; bzc[c] = bz_ln[lane + 32 * c]; }

  const float* zrow = z + ((size_t)i * N_SEQ + (size_t)jblk * 64) * CZ;
  const unsigned TILE_DW = 16 * CZ;                 // 2048 dwords = 8KB

  tdm_load_1d(lds_off(&zstage[0][0]), zrow, TILE_DW);   // prologue: tile 0

  #pragma unroll
  for (int t = 0; t < 4; ++t) {
    if (t < 3) {
      tdm_load_1d(lds_off(&zstage[(t + 1) & 1][0]), zrow + (size_t)(t + 1) * TILE_DW,
                  TILE_DW);
      __builtin_amdgcn_s_wait_tensorcnt(1);   // tile t landed, t+1 in flight
    } else {
      __builtin_amdgcn_s_wait_tensorcnt(0);
    }

    // --- LayerNorm: whole wave per row, conflict-free LDS access ---
    const float* stg = &zstage[t & 1][0];
    #pragma unroll
    for (int r = 0; r < 16; ++r) {
      const float* rowp = stg + r * CZ;
      float x0 = rowp[lane], x1 = rowp[lane + 32];
      float x2 = rowp[lane + 64], x3 = rowp[lane + 96];
      float s = x0 + x1 + x2 + x3;
      float ss = x0 * x0 + x1 * x1 + x2 * x2 + x3 * x3;
      #pragma unroll
      for (int m = 16; m >= 1; m >>= 1) {
        s += __shfl_xor(s, m, 32);
        ss += __shfl_xor(ss, m, 32);
      }
      float mean = s * (1.f / CZ);
      float inv = rsqrtf(ss * (1.f / CZ) - mean * mean + EPS);
      zt[r][lane]      = (_Float16)((x0 - mean) * inv * gzc[0] + bzc[0]);
      zt[r][lane + 32] = (_Float16)((x1 - mean) * inv * gzc[1] + bzc[1]);
      zt[r][lane + 64] = (_Float16)((x2 - mean) * inv * gzc[2] + bzc[2]);
      zt[r][lane + 96] = (_Float16)((x3 - mean) * inv * gzc[3] + bzc[3]);
    }
    __syncthreads();

    // --- 16x16 (j x h) projection: 4 chained K=32 WMMAs over C_Z=128 ---
    v8f acc = {};
    #pragma unroll
    for (int kstep = 0; kstep < 4; ++kstep) {
      v16h af = *(const v16h*)(&zt[lane & 15][kstep * 32 + ((lane >> 4) << 4)]);
      v16h bf = *(const v16h*)(WzT + (size_t)(lane & 15) * CZ + kstep * 32 +
                               ((lane >> 4) << 4));
      acc = wmma32(af, bf, acc);
    }
    int h = lane & 15;
    float bzv = bz[h];
    int j0 = jblk * 64 + t * 16;
    #pragma unroll
    for (int v = 0; v < 8; ++v) {
      int jr = j0 + v + ((lane >> 4) << 3);
      bias[((size_t)i * N_SEQ + jr) * NH + h] = acc[v] + bzv;  // coalesced over h
    }
    __syncthreads();
  }
}

// ---------------------------------------------------------------------------
// Kernel 5: repack q/k/v into per-head padded f16 layouts
//   qp,kp : [H][N][32]  (q pre-scaled by 1/sqrt(D));  vpT : [H][32][N]
// ---------------------------------------------------------------------------
__global__ void k_repack(const float* __restrict__ qkvg, _Float16* __restrict__ qp,
                         _Float16* __restrict__ kp, _Float16* __restrict__ vpT) {
  const float scale = 0.20412414523193154f;  // 1/sqrt(24)
  int total = NH * N_SEQ * DP;
  int stride = gridDim.x * blockDim.x;
  for (int idx = blockIdx.x * blockDim.x + threadIdx.x; idx < total; idx += stride) {
    int d = idx & (DP - 1);
    int n = (idx >> 5) & (N_SEQ - 1);
    int h = idx >> 14;                // 32*512 = 2^14 per head
    float qv = 0.f, kv = 0.f, vv = 0.f;
    if (d < DH) {
      int c = h * DH + d;
      qv = qkvg[(size_t)n * CS + c] * scale;
      kv = qkvg[(size_t)(N_SEQ * CS) + (size_t)n * CS + c];
      vv = qkvg[(size_t)(2 * N_SEQ * CS) + (size_t)n * CS + c];
    }
    qp[idx] = (_Float16)qv;
    kp[idx] = (_Float16)kv;
    vpT[((size_t)h * DP + d) * N_SEQ + n] = (_Float16)vv;
  }
}

// ---------------------------------------------------------------------------
// Kernel 6: flash attention per (head, 16-query tile); one wave per block.
// scores = q@k^T + bias ; streaming softmax ; o = p@v ; gate ; write f16.
// ---------------------------------------------------------------------------
__global__ void k_attn(const _Float16* __restrict__ qp, const _Float16* __restrict__ kp,
                       const _Float16* __restrict__ vpT, const float* __restrict__ bias,
                       const float* __restrict__ graw, const float* __restrict__ bg,
                       _Float16* __restrict__ o16) {
  int h = blockIdx.x;
  int q0 = blockIdx.y * 16;
  int lane = threadIdx.x;
  __shared__ _Float16 P[16][32];

  const _Float16* qh = qp + (size_t)h * N_SEQ * DP;
  const _Float16* kh = kp + (size_t)h * N_SEQ * DP;
  const _Float16* vh = vpT + (size_t)h * DP * N_SEQ;

  v16h aq = load_frag(qh, DP, q0, 0);          // q tile, loaded once
  float mrun[8], lrun[8];
  v8f o1 = {}, o2 = {};
  #pragma unroll
  for (int v = 0; v < 8; ++v) { mrun[v] = -1e30f; lrun[v] = 0.f; }

  int n = lane & 15;
  int rbase = (lane >> 4) << 3;

  for (int kk = 0; kk < N_SEQ; kk += 32) {
    v16h bk1 = load_frag(kh, DP, kk, 0);       // keys kk..kk+15 as B-frag
    v16h bk2 = load_frag(kh, DP, kk + 16, 0);  // keys kk+16..kk+31
    v8f s1 = {}, s2 = {};
    s1 = wmma32(aq, bk1, s1);
    s2 = wmma32(aq, bk2, s2);

    #pragma unroll
    for (int v = 0; v < 8; ++v) {
      size_t rq = (size_t)(q0 + v + rbase);
      s1[v] += bias[(rq * N_SEQ + kk + n) * NH + h];
      s2[v] += bias[(rq * N_SEQ + kk + 16 + n) * NH + h];
    }
    #pragma unroll
    for (int v = 0; v < 8; ++v) {
      float m = fmaxf(s1[v], s2[v]);
      #pragma unroll
      for (int msk = 1; msk <= 8; msk <<= 1) m = fmaxf(m, __shfl_xor(m, msk, 32));
      float mnew = fmaxf(mrun[v], m);
      float resc = __expf(mrun[v] - mnew);
      mrun[v] = mnew;
      lrun[v] *= resc;
      o1[v] *= resc;
      o2[v] *= resc;
      float p1 = __expf(s1[v] - mnew);
      float p2 = __expf(s2[v] - mnew);
      float ls = p1 + p2;
      #pragma unroll
      for (int msk = 1; msk <= 8; msk <<= 1) ls += __shfl_xor(ls, msk, 32);
      lrun[v] += ls;
      P[v + rbase][n] = (_Float16)p1;
      P[v + rbase][16 + n] = (_Float16)p2;
    }
    __syncthreads();
    v16h pa = *(const v16h*)(&P[lane & 15][(lane >> 4) << 4]);  // re-frag P via LDS
    v16h bv1 = load_frag(vh, N_SEQ, 0, kk);    // v cols d=0..15
    v16h bv2 = load_frag(vh, N_SEQ, 16, kk);   // v cols d=16..31 (pad)
    o1 = wmma32(pa, bv1, o1);
    o2 = wmma32(pa, bv2, o2);
    __syncthreads();
  }

  // epilogue: 1/l, gate = sigmoid(a_n@Wg + bg), write f16 [N][H*D]
  int d1 = n, d2 = 16 + n;
  #pragma unroll
  for (int v = 0; v < 8; ++v) {
    int row = q0 + v + rbase;
    float invl = 1.f / lrun[v];
    int c1 = h * DH + d1;
    float g1 = 1.f / (1.f + __expf(-(graw[(size_t)row * CS + c1] + bg[c1])));
    o16[(size_t)row * CS + c1] = (_Float16)(o1[v] * invl * g1);
    if (d2 < DH) {
      int c2 = h * DH + d2;
      float g2 = 1.f / (1.f + __expf(-(graw[(size_t)row * CS + c2] + bg[c2])));
      o16[(size_t)row * CS + c2] = (_Float16)(o2[v] * invl * g2);
    }
  }
}

// ---------------------------------------------------------------------------
// Host-side orchestration
// ---------------------------------------------------------------------------
extern "C" void kernel_launch(void* const* d_in, const int* in_sizes, int n_in,
                              void* d_out, int out_size, void* d_ws, size_t ws_size,
                              hipStream_t stream) {
  (void)in_sizes; (void)n_in; (void)out_size; (void)ws_size;
  const float* a   = (const float*)d_in[0];
  const float* z   = (const float*)d_in[1];
  const float* g_a = (const float*)d_in[2];
  const float* b_a = (const float*)d_in[3];
  const float* g_z = (const float*)d_in[4];
  const float* b_z = (const float*)d_in[5];
  const float* Wz  = (const float*)d_in[6];
  const float* bz  = (const float*)d_in[7];
  const float* Wq  = (const float*)d_in[8];
  const float* Wk  = (const float*)d_in[9];
  const float* Wv  = (const float*)d_in[10];
  const float* Wg  = (const float*)d_in[11];
  const float* bg  = (const float*)d_in[12];
  const float* Wo  = (const float*)d_in[13];
  const float* bo  = (const float*)d_in[14];
  float* out = (float*)d_out;

  char* w = (char*)d_ws;
  auto take = [&](size_t bytes) -> char* {
    char* p = w;
    w += (bytes + 255) & ~(size_t)255;
    return p;
  };
  _Float16* a_n16 = (_Float16*)take((size_t)N_SEQ * CS * 2);
  _Float16* WqT   = (_Float16*)take((size_t)CS * CS * 2);
  _Float16* WkT   = (_Float16*)take((size_t)CS * CS * 2);
  _Float16* WvT   = (_Float16*)take((size_t)CS * CS * 2);
  _Float16* WgT   = (_Float16*)take((size_t)CS * CS * 2);
  _Float16* WoT   = (_Float16*)take((size_t)CS * CS * 2);
  _Float16* WzT   = (_Float16*)take((size_t)NH * CZ * 2);
  float*    qkvg  = (float*)take((size_t)4 * N_SEQ * CS * 4);
  _Float16* qp    = (_Float16*)take((size_t)NH * N_SEQ * DP * 2);
  _Float16* kp    = (_Float16*)take((size_t)NH * N_SEQ * DP * 2);
  _Float16* vpT   = (_Float16*)take((size_t)NH * DP * N_SEQ * 2);
  float*    biasb = (float*)take((size_t)N_SEQ * N_SEQ * NH * 4);
  _Float16* o16   = (_Float16*)take((size_t)N_SEQ * CS * 2);

  // 1. LN(a) -> f16
  k_ln_a<<<dim3(N_SEQ), dim3(128), 0, stream>>>(a, g_a, b_a, a_n16);
  // 2. weights -> f16 transposed
  k_prep_w<<<dim3(512), dim3(256), 0, stream>>>(Wq, Wk, Wv, Wg, Wo, Wz,
                                                WqT, WkT, WvT, WgT, WoT, WzT);
  // 3. q/k/v/g projections (f16 WMMA, f32 accumulate)
  dim3 gg(N_SEQ / 16, CS / 64);
  k_gemm_f16<<<gg, dim3(128), 0, stream>>>(a_n16, WqT, nullptr,
                                           qkvg + 0 * N_SEQ * CS, N_SEQ, CS, CS);
  k_gemm_f16<<<gg, dim3(128), 0, stream>>>(a_n16, WkT, nullptr,
                                           qkvg + 1 * N_SEQ * CS, N_SEQ, CS, CS);
  k_gemm_f16<<<gg, dim3(128), 0, stream>>>(a_n16, WvT, nullptr,
                                           qkvg + 2 * N_SEQ * CS, N_SEQ, CS, CS);
  k_gemm_f16<<<gg, dim3(128), 0, stream>>>(a_n16, WgT, nullptr,
                                           qkvg + 3 * N_SEQ * CS, N_SEQ, CS, CS);
  // 4. fused LN(z) + pair-bias projection (TDM double-buffered HBM-bound pass)
  k_pair_bias<<<dim3(N_SEQ, N_SEQ / 64), dim3(32), 0, stream>>>(z, g_z, b_z, WzT,
                                                                bz, biasb);
  // 5. repack q/k/v into per-head padded layouts
  k_repack<<<dim3(1024), dim3(256), 0, stream>>>(qkvg, qp, kp, vpT);
  // 6. flash attention with bias + gating -> f16 [N][H*D]
  k_attn<<<dim3(NH, N_SEQ / 16), dim3(32), 0, stream>>>(
      qp, kp, vpT, biasb, qkvg + (size_t)3 * N_SEQ * CS, bg, o16);
  // 7. output projection + bo -> d_out (f32)
  k_gemm_f16<<<gg, dim3(128), 0, stream>>>(o16, WoT, bo, out, N_SEQ, CS, CS);
}
